// SwinTransformerBlock_50397146251601
// MI455X (gfx1250) — matrix operand order repaired and checked by reading, not compiled, not run on hardware
//
#include <hip/hip_runtime.h>
#include <math.h>

// ---------------------------------------------------------------------------
// Swin-V2-CR block for MI455X (gfx1250, wave32, WMMA).
// All heavy GEMMs use v_wmma_f32_16x16x32_f16 with fp32 accumulation.
// Dims: B=4, H=W=256, C=180 (pad 192), heads=6, hd=30 (pad 32), ws=8, Lw=64.
//
// Roofline: ~216 GFLOP of GEMM work vs >=600 MB activation traffic; at
// 23.3 TB/s the block is bandwidth-bound, so intermediates are f16 and the
// MLP hidden tensor (400 MB if materialized) is kept entirely in LDS.
// ---------------------------------------------------------------------------

typedef __attribute__((ext_vector_type(16))) _Float16 v16h;
typedef __attribute__((ext_vector_type(8)))  _Float16 v8h;
typedef __attribute__((ext_vector_type(8)))  float    v8f;

#define T_TOK   262144      // total tokens = 4*256*256
#define NWIN    4096        // total windows = 4*(32*32)
#define CP      192         // padded channel dim
#define QKVN    576         // padded qkv output cols (3 * 6 heads * 32)
#define HIDP    768         // padded MLP hidden (720 -> 768)

static __device__ __forceinline__ v16h mk16(v8h lo, v8h hi) {
    v16h r;
#pragma unroll
    for (int i = 0; i < 8; ++i) { r[i] = lo[i]; r[8 + i] = hi[i]; }
    return r;
}

// A fragment (16x32 f16): lane L holds row m = m0 + (L&15); halves 0..15 at
// K = k0 + 16*(L>>4) .. +15 contiguous.
static __device__ __forceinline__ v16h load_afrag(const _Float16* base, int stride,
                                                  int m0, int k0, int lane) {
    const _Float16* p = base + (size_t)(m0 + (lane & 15)) * stride + k0 + 16 * (lane >> 4);
    v8h lo = *(const v8h*)p;
    v8h hi = *(const v8h*)(p + 8);
    return mk16(lo, hi);
}

// B fragment (32x16 f16) from an [N][K] ("Bt") layout: lane L holds column
// n = n0 + (L&15), K = k0 + 16*(L>>4) .. +15 contiguous -> one 32-byte load.
static __device__ __forceinline__ v16h load_bfrag(const _Float16* bt, int stride,
                                                  int n0, int k0, int lane) {
    return *(const v16h*)(bt + (size_t)(n0 + (lane & 15)) * stride + k0 + 16 * (lane >> 4));
}

static __device__ __forceinline__ v8f wmma_f16(v16h a, v16h b, v8f c) {
    return __builtin_amdgcn_wmma_f32_16x16x32_f16(false, a, false, b, (short)0, c,
                                                  false, false);
}

// ---------------------------------------------------------------------------
// Weight prep: transpose + pad fp32 [srcK][srcN] -> f16 [dstN][dstK], plus bias.
// ---------------------------------------------------------------------------
__global__ void k_tpad(const float* __restrict__ src, int srcK, int srcN,
                       _Float16* __restrict__ dst, int dstN, int dstK,
                       const float* __restrict__ bsrc, float* __restrict__ bdst) {
    int i = blockIdx.x * 256 + threadIdx.x;
    int total = dstN * dstK;
    if (i < total) {
        int n = i / dstK, k = i % dstK;
        float v = (n < srcN && k < srcK) ? src[k * srcN + n] : 0.0f;
        dst[i] = (_Float16)v;
    }
    if (i < dstN) bdst[i] = (i < srcN) ? bsrc[i] : 0.0f;
}

// qkv_w (180x540) -> [576][192] f16, columns remapped to {q,k,v} x head-major
// padded-32 layout: col' = part*192 + head*32 + d  (d<30 valid).
__global__ void k_tpad_qkv2(const float* __restrict__ src, const float* __restrict__ bsrc,
                            _Float16* __restrict__ dst, float* __restrict__ bdst) {
    int i = blockIdx.x * 256 + threadIdx.x;
    if (i < QKVN * CP) {
        int n = i / CP, k = i % CP;
        int part = n / 192, rem = n % 192;
        int head = rem / 32, d = rem % 32;
        float v = 0.0f;
        if (d < 30 && k < 180) v = src[k * 540 + part * 180 + head * 30 + d];
        dst[i] = (_Float16)v;
    }
    if (i < QKVN) {
        int part = i / 192, rem = i % 192;
        int head = rem / 32, d = rem % 32;
        bdst[i] = (d < 30) ? bsrc[part * 180 + head * 30 + d] : 0.0f;
    }
}

// ---------------------------------------------------------------------------
// Relative-position meta-MLP bias table: bias6[6][64][64] fp32.
// ---------------------------------------------------------------------------
__global__ void k_metabias(const float* __restrict__ w1, const float* __restrict__ b1,
                           const float* __restrict__ w2, const float* __restrict__ b2,
                           float* __restrict__ bias6) {
    int idx = blockIdx.x * 256 + threadIdx.x;          // 4096 pairs
    int a = idx >> 6, bb = idx & 63;
    float di = (float)((a >> 3) - (bb >> 3));
    float dj = (float)((a & 7) - (bb & 7));
    float s0 = (di > 0.f) ? 1.f : ((di < 0.f) ? -1.f : 0.f);
    float s1 = (dj > 0.f) ? 1.f : ((dj < 0.f) ? -1.f : 0.f);
    float c0 = s0 * log1pf(fabsf(di));
    float c1 = s1 * log1pf(fabsf(dj));
    float acc[6] = {0.f, 0.f, 0.f, 0.f, 0.f, 0.f};
    for (int m = 0; m < 384; ++m) {
        float t = fmaxf(c0 * w1[m] + c1 * w1[384 + m] + b1[m], 0.0f);
#pragma unroll
        for (int h = 0; h < 6; ++h) acc[h] += t * w2[m * 6 + h];
    }
#pragma unroll
    for (int h = 0; h < 6; ++h) bias6[h * 4096 + idx] = acc[h] + b2[h];
}

// ---------------------------------------------------------------------------
// LN1 + cyclic shift(-4,-4) + window partition -> xw f16 [NWIN*64][192].
// ---------------------------------------------------------------------------
__global__ void k_ln1(const float* __restrict__ x, const float* __restrict__ nw,
                      const float* __restrict__ nb, _Float16* __restrict__ xw) {
    int win = blockIdx.x;
    int t   = threadIdx.x;                              // 64 threads
    int bi  = win >> 10, wrem = win & 1023;
    int wy  = wrem >> 5, wx = wrem & 31;
    int i   = t >> 3, j = t & 7;
    int h   = (wy * 8 + i + 4) & 255;
    int w   = (wx * 8 + j + 4) & 255;
    const float* row = x + ((size_t)bi * 65536 + h * 256 + w) * 180;
    float mean = 0.f;
    for (int c = 0; c < 180; ++c) mean += row[c];
    mean *= (1.0f / 180.0f);
    float var = 0.f;
    for (int c = 0; c < 180; ++c) { float d = row[c] - mean; var += d * d; }
    var *= (1.0f / 180.0f);
    float rinv = rsqrtf(var + 1e-5f);
    _Float16* o = xw + (size_t)(win * 64 + t) * CP;
    for (int c = 0; c < 180; ++c) o[c] = (_Float16)((row[c] - mean) * rinv * nw[c] + nb[c]);
    for (int c = 180; c < CP; ++c) o[c] = (_Float16)0.0f;
}

// ---------------------------------------------------------------------------
// QKV GEMM: [T][192] @ Wt[576][192] -> qkv f16 [T][576] (+bias).
// Block = 8 waves, tile 128(M) x 64(N), K=192 (6 WMMA K-steps).
// ---------------------------------------------------------------------------
__global__ __launch_bounds__(256) void k_gemm_qkv(const _Float16* __restrict__ A,
                                                  const _Float16* __restrict__ Bt,
                                                  const float* __restrict__ bias,
                                                  _Float16* __restrict__ O) {
    __shared__ __align__(128) _Float16 As[128][CP];
    int tid = threadIdx.x;
    int rowBlk = blockIdx.x * 128;
    int colBlk = blockIdx.y * 64;
    {
        const v8h* src = (const v8h*)(A + (size_t)rowBlk * CP);
        v8h* dst = (v8h*)&As[0][0];
#pragma unroll
        for (int i = 0; i < 12; ++i) dst[tid + i * 256] = src[tid + i * 256];
    }
    __builtin_prefetch(Bt + (size_t)colBlk * CP, 0, 1);
    __syncthreads();
    int wave = tid >> 5, lane = tid & 31;
    int m0 = wave * 16;
    v8f acc0 = {}, acc1 = {}, acc2 = {}, acc3 = {};
    for (int k0 = 0; k0 < CP; k0 += 32) {
        v16h a = load_afrag(&As[0][0], CP, m0, k0, lane);
        acc0 = wmma_f16(a, load_bfrag(Bt, CP, colBlk +  0, k0, lane), acc0);
        acc1 = wmma_f16(a, load_bfrag(Bt, CP, colBlk + 16, k0, lane), acc1);
        acc2 = wmma_f16(a, load_bfrag(Bt, CP, colBlk + 32, k0, lane), acc2);
        acc3 = wmma_f16(a, load_bfrag(Bt, CP, colBlk + 48, k0, lane), acc3);
    }
    int ln = lane & 15, hl = lane >> 4;
    v8f accs[4] = {acc0, acc1, acc2, acc3};
#pragma unroll
    for (int tt = 0; tt < 4; ++tt) {
        int n = colBlk + tt * 16 + ln;
        float bn = bias[n];
#pragma unroll
        for (int r = 0; r < 8; ++r) {
            int m = rowBlk + m0 + 8 * hl + r;
            O[(size_t)m * QKVN + n] = (_Float16)(accs[tt][r] + bn);
        }
    }
}

// ---------------------------------------------------------------------------
// Cosine-attention q/k normalization (+ fold per-head logit scale into q).
// ---------------------------------------------------------------------------
__global__ void k_qknorm(_Float16* __restrict__ qkv, const float* __restrict__ ls) {
    int gid = blockIdx.x * 256 + threadIdx.x;           // T_TOK * 6 exact
    int h = gid % 6;
    size_t tok = (size_t)(gid / 6);
    _Float16* q = qkv + tok * QKVN + h * 32;
    _Float16* k = q + 192;
    float s = __expf(fminf(ls[h], 4.60517019f));        // log(100)
    float nq = 0.f, nk = 0.f;
    for (int d = 0; d < 30; ++d) {
        float a = (float)q[d]; nq += a * a;
        float c = (float)k[d]; nk += c * c;
    }
    float iq = s / fmaxf(sqrtf(nq), 1e-12f);
    float ik = 1.0f / fmaxf(sqrtf(nk), 1e-12f);
    for (int d = 0; d < 30; ++d) {
        q[d] = (_Float16)((float)q[d] * iq);
        k[d] = (_Float16)((float)k[d] * ik);
    }
}

static __device__ __forceinline__ int regionOf(int tok, int wy, int wx) {
    int i = tok >> 3, j = tok & 7;
    int hs = wy * 8 + i, ws_ = wx * 8 + j;
    int rh = (hs < 248) ? 0 : ((hs < 252) ? 1 : 2);
    int rw = (ws_ < 248) ? 0 : ((ws_ < 252) ? 1 : 2);
    return rh * 3 + rw;
}

// ---------------------------------------------------------------------------
// Attention per window: S = qn@knT (WMMA, K=32), +bias+mask, fp32 softmax via
// wave32 shuffles, P->LDS, Vt->LDS, O = P@V (WMMA).  Block = 4 waves.
// ---------------------------------------------------------------------------
__global__ __launch_bounds__(128) void k_attn(const _Float16* __restrict__ QKV,
                                              const float* __restrict__ bias6,
                                              _Float16* __restrict__ Oo) {
    __shared__ __align__(128) _Float16 Ps[64][64];
    __shared__ __align__(128) _Float16 Vs[32][64];
    int win = blockIdx.x;
    int wrem = win & 1023, wy = wrem >> 5, wx = wrem & 31;
    int tid = threadIdx.x, wave = tid >> 5, lane = tid & 31;
    int m0 = wave * 16, ln = lane & 15, hl = lane >> 4;
    const _Float16* base = QKV + (size_t)win * 64 * QKVN;

    // zero the 180..191 channel pad of this window's attention output
    for (int i = tid; i < 64 * 12; i += 128) {
        int m = i / 12, c = 180 + i % 12;
        Oo[(size_t)(win * 64 + m) * CP + c] = (_Float16)0.0f;
    }

    for (int h = 0; h < 6; ++h) {
        // stage V transposed: Vs[d][t] = V[t][d]
        for (int i = tid; i < 2048; i += 128) {
            int d = i >> 6, t = i & 63;
            Vs[d][t] = base[(size_t)t * QKVN + 384 + h * 32 + d];
        }
        // S = qn @ knT  (single K-step of 32 per tile)
        v16h a = load_afrag(base, QKVN, m0, h * 32, lane);
        v8f s[4];
#pragma unroll
        for (int tt = 0; tt < 4; ++tt) {
            v8f z = {};
            v16h b = load_bfrag(base, QKVN, tt * 16, 192 + h * 32, lane);
            s[tt] = wmma_f16(a, b, z);
        }
        // bias + shift-mask + softmax (rows live in lane-halves: m=8*hl+r)
        const float* bh = bias6 + h * 4096;
        int rn[4];
#pragma unroll
        for (int tt = 0; tt < 4; ++tt) rn[tt] = regionOf(tt * 16 + ln, wy, wx);
#pragma unroll
        for (int r = 0; r < 8; ++r) {
            int m = m0 + 8 * hl + r;
            int rm = regionOf(m, wy, wx);
            float mx = -1e30f;
#pragma unroll
            for (int tt = 0; tt < 4; ++tt) {
                float v = s[tt][r] + bh[m * 64 + tt * 16 + ln];
                if (rm != rn[tt]) v -= 100.0f;
                s[tt][r] = v;
                mx = fmaxf(mx, v);
            }
            for (int off = 8; off >= 1; off >>= 1) mx = fmaxf(mx, __shfl_xor(mx, off, 32));
            float sum = 0.f;
#pragma unroll
            for (int tt = 0; tt < 4; ++tt) {
                float e = __expf(s[tt][r] - mx);
                s[tt][r] = e;
                sum += e;
            }
            for (int off = 8; off >= 1; off >>= 1) sum += __shfl_xor(sum, off, 32);
            float inv = 1.0f / sum;
#pragma unroll
            for (int tt = 0; tt < 4; ++tt)
                Ps[m][tt * 16 + ln] = (_Float16)(s[tt][r] * inv);
        }
        __syncthreads();
        // O = P @ V : M=64, N=32, K=64
        v8f o0 = {}, o1 = {};
        for (int k0 = 0; k0 < 64; k0 += 32) {
            v16h pa = load_afrag(&Ps[0][0], 64, m0, k0, lane);
            o0 = wmma_f16(pa, load_bfrag(&Vs[0][0], 64, 0,  k0, lane), o0);
            o1 = wmma_f16(pa, load_bfrag(&Vs[0][0], 64, 16, k0, lane), o1);
        }
        v8f os[2] = {o0, o1};
#pragma unroll
        for (int tt = 0; tt < 2; ++tt) {
            int n = tt * 16 + ln;
            if (n < 30) {
#pragma unroll
                for (int r = 0; r < 8; ++r) {
                    int m = m0 + 8 * hl + r;
                    Oo[(size_t)(win * 64 + m) * CP + h * 30 + n] = (_Float16)os[tt][r];
                }
            }
        }
        __syncthreads();
    }
}

// ---------------------------------------------------------------------------
// Proj GEMM fused with window-reverse + un-shift + residual into d_out (fp32).
// ---------------------------------------------------------------------------
__global__ __launch_bounds__(256) void k_gemm_proj(const _Float16* __restrict__ A,
                                                   const _Float16* __restrict__ Bt,
                                                   const float* __restrict__ bias,
                                                   const float* __restrict__ xin,
                                                   float* __restrict__ out) {
    __shared__ __align__(128) _Float16 As[128][CP];
    int tid = threadIdx.x;
    int rowBlk = blockIdx.x * 128;
    int colBlk = blockIdx.y * 64;
    {
        const v8h* src = (const v8h*)(A + (size_t)rowBlk * CP);
        v8h* dst = (v8h*)&As[0][0];
#pragma unroll
        for (int i = 0; i < 12; ++i) dst[tid + i * 256] = src[tid + i * 256];
    }
    __builtin_prefetch(Bt + (size_t)colBlk * CP, 0, 1);
    __syncthreads();
    int wave = tid >> 5, lane = tid & 31;
    int m0 = wave * 16, ln = lane & 15, hl = lane >> 4;
    v8f acc[4] = {{}, {}, {}, {}};
    for (int k0 = 0; k0 < CP; k0 += 32) {
        v16h a = load_afrag(&As[0][0], CP, m0, k0, lane);
#pragma unroll
        for (int tt = 0; tt < 4; ++tt)
            acc[tt] = wmma_f16(a, load_bfrag(Bt, CP, colBlk + tt * 16, k0, lane), acc[tt]);
    }
#pragma unroll
    for (int tt = 0; tt < 4; ++tt) {
        int col = colBlk + tt * 16 + ln;
        if (col < 180) {
            float bn = bias[col];
#pragma unroll
            for (int r = 0; r < 8; ++r) {
                int tok = rowBlk + m0 + 8 * hl + r;
                int win = tok >> 6, t = tok & 63;
                int bi = win >> 10, wrem = win & 1023;
                int wy = wrem >> 5, wx = wrem & 31;
                int hs = wy * 8 + (t >> 3), ws_ = wx * 8 + (t & 7);
                int hh = (hs + 4) & 255, ww = (ws_ + 4) & 255;
                size_t idx = ((size_t)bi * 65536 + hh * 256 + ww) * 180 + col;
                out[idx] = xin[idx] + acc[tt][r] + bn;
            }
        }
    }
}

// ---------------------------------------------------------------------------
// Fused LN2 + MLP (C -> 720 (GELU) -> C) with residual; hidden lives in LDS
// in chunks of 256 cols so the 400 MB hidden tensor never touches HBM.
// Block = 8 waves, 64 tokens.
// ---------------------------------------------------------------------------
__global__ __launch_bounds__(256) void k_mlp(const float* __restrict__ n2w,
                                             const float* __restrict__ n2b,
                                             const _Float16* __restrict__ W1t,
                                             const float* __restrict__ B1,
                                             const _Float16* __restrict__ W2t,
                                             const float* __restrict__ B2,
                                             float* __restrict__ out) {
    __shared__ __align__(128) _Float16 Hs[64][CP];     // LN2(x1) f16
    __shared__ __align__(128) _Float16 Hh[64][256];    // hidden chunk f16
    int tid = threadIdx.x;
    int tokBlk = blockIdx.x * 64;
    if (tid < 64) {
        const float* row = out + (size_t)(tokBlk + tid) * 180;
        float mean = 0.f;
        for (int c = 0; c < 180; ++c) mean += row[c];
        mean *= (1.0f / 180.0f);
        float var = 0.f;
        for (int c = 0; c < 180; ++c) { float d = row[c] - mean; var += d * d; }
        var *= (1.0f / 180.0f);
        float rinv = rsqrtf(var + 1e-5f);
        for (int c = 0; c < 180; ++c)
            Hs[tid][c] = (_Float16)((row[c] - mean) * rinv * n2w[c] + n2b[c]);
        for (int c = 180; c < CP; ++c) Hs[tid][c] = (_Float16)0.0f;
    }
    __syncthreads();
    int wave = tid >> 5, lane = tid & 31;
    int mw = wave & 3, nwv = wave >> 2;
    int m0 = mw * 16, ln = lane & 15, hl = lane >> 4;

    v8f acc2[6] = {{}, {}, {}, {}, {}, {}};            // final 16x96 per wave
    for (int chunk = 0; chunk < 3; ++chunk) {
        // GEMM1 chunk: hidden cols [chunk*256, chunk*256+256) , GELU -> Hh
#pragma unroll
        for (int tt = 0; tt < 8; ++tt) {
            int nloc = nwv * 128 + tt * 16;
            int nglob = chunk * 256 + nloc;
            v8f acc = {};
            for (int k0 = 0; k0 < CP; k0 += 32) {
                v16h a = load_afrag(&Hs[0][0], CP, m0, k0, lane);
                acc = wmma_f16(a, load_bfrag(W1t, CP, nglob, k0, lane), acc);
            }
            int n = nglob + ln;
            float bn = B1[n];
#pragma unroll
            for (int r = 0; r < 8; ++r) {
                int m = m0 + 8 * hl + r;
                float v = acc[r] + bn;
                float g = 0.5f * v * (1.0f + erff(v * 0.70710678f));
                Hh[m][nloc + ln] = (_Float16)g;
            }
        }
        __syncthreads();
        // GEMM2 partial: accumulate this K-chunk of 256 into acc2
#pragma unroll
        for (int tt = 0; tt < 6; ++tt) {
            int n0 = nwv * 96 + tt * 16;
            for (int k0 = 0; k0 < 256; k0 += 32) {
                v16h a = load_afrag(&Hh[0][0], 256, m0, k0, lane);
                acc2[tt] = wmma_f16(a, load_bfrag(W2t, HIDP, n0, chunk * 256 + k0, lane),
                                    acc2[tt]);
            }
        }
        __syncthreads();
    }
    // epilogue: residual add into d_out
#pragma unroll
    for (int tt = 0; tt < 6; ++tt) {
        int n = nwv * 96 + tt * 16 + ln;
        if (n < 180) {
            float bn = B2[n];
#pragma unroll
            for (int r = 0; r < 8; ++r) {
                int m = m0 + 8 * hl + r;
                size_t idx = (size_t)(tokBlk + m) * 180 + n;
                out[idx] = out[idx] + acc2[tt][r] + bn;
            }
        }
    }
}

// ---------------------------------------------------------------------------
extern "C" void kernel_launch(void* const* d_in, const int* in_sizes, int n_in,
                              void* d_out, int out_size, void* d_ws, size_t ws_size,
                              hipStream_t stream) {
    const float* x       = (const float*)d_in[0];
    const float* norm1_w = (const float*)d_in[3];
    const float* norm1_b = (const float*)d_in[4];
    const float* qkv_w   = (const float*)d_in[5];
    const float* qkv_b   = (const float*)d_in[6];
    const float* lscale  = (const float*)d_in[7];
    const float* mw1     = (const float*)d_in[8];
    const float* mb1     = (const float*)d_in[9];
    const float* mw2     = (const float*)d_in[10];
    const float* mb2     = (const float*)d_in[11];
    const float* proj_w  = (const float*)d_in[12];
    const float* proj_b  = (const float*)d_in[13];
    const float* norm2_w = (const float*)d_in[14];
    const float* norm2_b = (const float*)d_in[15];
    const float* fc1_w   = (const float*)d_in[16];
    const float* fc1_b   = (const float*)d_in[17];
    const float* fc2_w   = (const float*)d_in[18];
    const float* fc2_b   = (const float*)d_in[19];
    float* out = (float*)d_out;

    // workspace carve (256-byte aligned)
    size_t off = 0;
    auto carve = [&](size_t bytes) -> void* {
        void* p = (char*)d_ws + off;
        off += (bytes + 255) & ~(size_t)255;
        return p;
    };
    _Float16* xw    = (_Float16*)carve((size_t)T_TOK * CP * 2);    // reused as attn-out
    _Float16* qkv   = (_Float16*)carve((size_t)T_TOK * QKVN * 2);
    float*    bias6 = (float*)carve(6 * 4096 * 4);
    _Float16* qkvWt = (_Float16*)carve((size_t)QKVN * CP * 2);
    float*    qkvB  = (float*)carve(QKVN * 4);
    _Float16* projWt= (_Float16*)carve((size_t)CP * CP * 2);
    float*    projB = (float*)carve(CP * 4);
    _Float16* fc1Wt = (_Float16*)carve((size_t)HIDP * CP * 2);
    float*    fc1B  = (float*)carve(HIDP * 4);
    _Float16* fc2Wt = (_Float16*)carve((size_t)CP * HIDP * 2);
    float*    fc2B  = (float*)carve(CP * 4);
    _Float16* attno = xw;  // attnout reuses xw (QKV GEMM consumed xw already)

    // --- weight prep + bias table (tiny) ---
    k_tpad_qkv2<<<(QKVN * CP + 255) / 256, 256, 0, stream>>>(qkv_w, qkv_b, qkvWt, qkvB);
    k_tpad<<<(CP * CP + 255) / 256, 256, 0, stream>>>(proj_w, 180, 180, projWt, CP, CP,
                                                      proj_b, projB);
    k_tpad<<<(HIDP * CP + 255) / 256, 256, 0, stream>>>(fc1_w, 180, 720, fc1Wt, HIDP, CP,
                                                        fc1_b, fc1B);
    k_tpad<<<(CP * HIDP + 255) / 256, 256, 0, stream>>>(fc2_w, 720, 180, fc2Wt, CP, HIDP,
                                                        fc2_b, fc2B);
    k_metabias<<<4096 / 256, 256, 0, stream>>>(mw1, mb1, mw2, mb2, bias6);

    // --- main pipeline ---
    k_ln1<<<NWIN, 64, 0, stream>>>(x, norm1_w, norm1_b, xw);
    k_gemm_qkv<<<dim3(T_TOK / 128, QKVN / 64), 256, 0, stream>>>(xw, qkvWt, qkvB, qkv);
    k_qknorm<<<(T_TOK * 6) / 256, 256, 0, stream>>>(qkv, lscale);
    k_attn<<<NWIN, 128, 0, stream>>>(qkv, bias6, attno);
    k_gemm_proj<<<dim3(T_TOK / 128, CP / 64), 256, 0, stream>>>(attno, projWt, projB, x, out);
    k_mlp<<<T_TOK / 64, 256, 0, stream>>>(norm2_w, norm2_b, fc1Wt, fc1B, fc2Wt, fc2B, out);
}
